// RetroModel_30159260353087
// MI455X (gfx1250) — matrix operand order repaired and checked by reading, not compiled, not run on hardware
//
#include <hip/hip_runtime.h>
#include <hip/hip_bf16.h>
#include <math.h>

// ---------------------------------------------------------------------------
// RETRO-style transformer forward for MI455X (gfx1250), wave32 + WMMA.
// Dense GEMMs: v_wmma_f32_16x16x32_bf16, f32->bf16 staged through double-
// buffered LDS with vectorized (b128) global loads. Attention softmax / LN /
// RoPE are f32 VALU (small FLOP fraction; K/V reuse lives in the 192MB L2).
// ---------------------------------------------------------------------------

#define Bn   2
#define Sn   1024
#define Dn   1024
#define Hn   16
#define DKn  64
#define Ln   6
#define Vn   32000
#define DFFn 4096
#define CHUNKSn 16
#define NEIGHn  2
#define NLENn   128
#define CLn     64
#define CA_LAYERn 3

typedef __attribute__((ext_vector_type(16))) __bf16 bf16x16;
typedef __attribute__((ext_vector_type(8)))  __bf16 bf16x8;
typedef __attribute__((ext_vector_type(4)))  __bf16 bf16x4;
typedef __attribute__((ext_vector_type(8)))  float  f32x8;

__device__ __forceinline__ bf16x16 frag16(const __bf16* p0, const __bf16* p1) {
  bf16x8 lo = *(const bf16x8*)p0;
  bf16x8 hi = *(const bf16x8*)p1;
  return __builtin_shufflevector(lo, hi, 0,1,2,3,4,5,6,7,8,9,10,11,12,13,14,15);
}

// ------------------------------ WMMA GEMM ----------------------------------
// C[M,N] = A[M,K] @ B[K,N] + bias + (relu?) + residual.  Row-major all.
// Requires M%64==0, N%128==0, K%32==0 (true for every call site here).
#define BM 64
#define BN 128
#define BK 32
#define LDK (BK + 8)     // 80B row stride: b64/b128-aligned, bank-staggered

__global__ __launch_bounds__(256)
void k_gemm_bf16(const float* __restrict__ A, const float* __restrict__ Bm,
                 const float* __restrict__ bias, const float* __restrict__ Res,
                 float* __restrict__ C, int M, int N, int K, int relu)
{
  __shared__ __bf16 As[2][BM][LDK];
  __shared__ __bf16 Bs[2][BN][LDK];   // B staged transposed: Bs[buf][n][k]

  const int tid  = threadIdx.x;
  const int lane = tid & 31;
  const int wave = tid >> 5;
  const int m0 = blockIdx.y * BM;
  const int n0 = blockIdx.x * BN;
  const int wm = (wave >> 2) * 32;     // wave row   (0|32)
  const int wn = (wave & 3)  * 32;     // wave col   (0..96)
  const int lh  = lane & 15;
  const int hiA = (lane >> 4) * 8;     // A frag K-base per ISA layout
  const int hiB = (lane >> 4) * 8;     // B frag K-base per ISA layout

  // per-thread staging coordinates (float4 granularity)
  const int arow0 = tid >> 3;                // A: rows, 8 float4s per row
  const int akk   = (tid & 7) * 4;
  const int bkk0  = tid >> 5;                // B: 32 float4s per K-row of 128
  const int bn4   = (tid & 31) * 4;

  f32x8 acc[2][2] = {};
  float4 aL[2], bL[4];

  // phase 1: issue all global loads for K-step k0 into registers (b128 each)
  auto load_g = [&](int k0) {
#pragma unroll
    for (int t = 0; t < 2; ++t)
      aL[t] = *(const float4*)&A[(size_t)(m0 + arow0 + t * 32) * K + k0 + akk];
#pragma unroll
    for (int t = 0; t < 4; ++t)
      bL[t] = *(const float4*)&Bm[(size_t)(k0 + bkk0 + t * 8) * N + n0 + bn4];
  };
  // phase 2: convert (v_cvt_pk_bf16_f32) and store to LDS buffer `buf`
  auto store_l = [&](int buf) {
#pragma unroll
    for (int t = 0; t < 2; ++t) {
      bf16x4 v;
      v.x = (__bf16)aL[t].x; v.y = (__bf16)aL[t].y;
      v.z = (__bf16)aL[t].z; v.w = (__bf16)aL[t].w;
      *(bf16x4*)&As[buf][arow0 + t * 32][akk] = v;      // ds_store_b64
    }
#pragma unroll
    for (int t = 0; t < 4; ++t) {                        // transposed scatter
      int kk = bkk0 + t * 8;
      Bs[buf][bn4 + 0][kk] = (__bf16)bL[t].x;
      Bs[buf][bn4 + 1][kk] = (__bf16)bL[t].y;
      Bs[buf][bn4 + 2][kk] = (__bf16)bL[t].z;
      Bs[buf][bn4 + 3][kk] = (__bf16)bL[t].w;
    }
  };

  load_g(0);
  store_l(0);
  __syncthreads();

  int buf = 0;
  for (int k0 = 0; k0 < K; k0 += BK) {
    const bool more = (k0 + BK) < K;
    if (more) {
      load_g(k0 + BK);                       // loads in flight under the math
      if (k0 + 2 * BK < K)                   // global_prefetch_b8 one more out
        __builtin_prefetch(&A[(size_t)(m0 + arow0) * K + k0 + 2 * BK], 0, 1);
    }

    bf16x16 a0 = frag16(&As[buf][wm + lh][hiA],      &As[buf][wm + lh][hiA + 16]);
    bf16x16 a1 = frag16(&As[buf][wm + 16 + lh][hiA], &As[buf][wm + 16 + lh][hiA + 16]);
    bf16x16 b0 = frag16(&Bs[buf][wn + lh][hiB],      &Bs[buf][wn + lh][hiB + 16]);
    bf16x16 b1 = frag16(&Bs[buf][wn + 16 + lh][hiB], &Bs[buf][wn + 16 + lh][hiB + 16]);

    acc[0][0] = __builtin_amdgcn_wmma_f32_16x16x32_bf16(false, a0, false, b0, (short)0, acc[0][0], false, false);
    acc[0][1] = __builtin_amdgcn_wmma_f32_16x16x32_bf16(false, a0, false, b1, (short)0, acc[0][1], false, false);
    acc[1][0] = __builtin_amdgcn_wmma_f32_16x16x32_bf16(false, a1, false, b0, (short)0, acc[1][0], false, false);
    acc[1][1] = __builtin_amdgcn_wmma_f32_16x16x32_bf16(false, a1, false, b1, (short)0, acc[1][1], false, false);

    if (more) store_l(buf ^ 1);              // fill the other buffer
    __syncthreads();
    buf ^= 1;
  }

  const int rb = (lane >> 4) * 8;            // C/D row sub-block per ISA layout
#pragma unroll
  for (int i = 0; i < 2; ++i) {
#pragma unroll
    for (int j = 0; j < 2; ++j) {
      int col = n0 + wn + j * 16 + lh;
      float bval = bias ? bias[col] : 0.0f;
#pragma unroll
      for (int r = 0; r < 8; ++r) {
        int row = m0 + wm + i * 16 + rb + r;
        float val = acc[i][j][r] + bval;
        if (relu) val = fmaxf(val, 0.0f);
        if (Res)  val += Res[(size_t)row * N + col];
        C[(size_t)row * N + col] = val;
      }
    }
  }
}

// ------------------------------- LayerNorm ---------------------------------
__global__ __launch_bounds__(256)
void k_ln(const float* __restrict__ in, float* __restrict__ out,
          const float* __restrict__ sc, const float* __restrict__ bs)
{
  const int row = blockIdx.x;
  const int tid = threadIdx.x;
  const float* x = in + (size_t)row * Dn;
  float v[4]; float s1 = 0.f, s2 = 0.f;
#pragma unroll
  for (int t = 0; t < 4; ++t) { float f = x[tid + t * 256]; v[t] = f; s1 += f; s2 += f * f; }
  __shared__ float r1[256], r2[256];
  r1[tid] = s1; r2[tid] = s2; __syncthreads();
  for (int off = 128; off > 0; off >>= 1) {
    if (tid < off) { r1[tid] += r1[tid + off]; r2[tid] += r2[tid + off]; }
    __syncthreads();
  }
  float mu  = r1[0] * (1.0f / Dn);
  float var = r2[0] * (1.0f / Dn) - mu * mu;
  float inv = rsqrtf(var + 1e-5f);
  float* o = out + (size_t)row * Dn;
#pragma unroll
  for (int t = 0; t < 4; ++t) {
    int c = tid + t * 256;
    o[c] = (v[t] - mu) * inv * sc[c] + bs[c];
  }
}

// LN of h[b, r+63] -> hs_pad[b, r] for r<961, zeros for r>=961 (CCA input).
__global__ __launch_bounds__(256)
void k_cca_ln_pad(const float* __restrict__ h, float* __restrict__ out,
                  const float* __restrict__ sc, const float* __restrict__ bs)
{
  const int r = blockIdx.x & 1023;
  const int b = blockIdx.x >> 10;
  const int tid = threadIdx.x;
  float* o = out + ((size_t)(b * 1024 + r)) * Dn;
  if (r >= Sn - (CLn - 1)) {            // padded rows: zeros (post-LN pad, as ref)
#pragma unroll
    for (int t = 0; t < 4; ++t) o[tid + t * 256] = 0.0f;
    return;
  }
  const float* x = h + ((size_t)(b * Sn + r + CLn - 1)) * Dn;
  float v[4]; float s1 = 0.f, s2 = 0.f;
#pragma unroll
  for (int t = 0; t < 4; ++t) { float f = x[tid + t * 256]; v[t] = f; s1 += f; s2 += f * f; }
  __shared__ float r1[256], r2[256];
  r1[tid] = s1; r2[tid] = s2; __syncthreads();
  for (int off = 128; off > 0; off >>= 1) {
    if (tid < off) { r1[tid] += r1[tid + off]; r2[tid] += r2[tid + off]; }
    __syncthreads();
  }
  float mu  = r1[0] * (1.0f / Dn);
  float var = r2[0] * (1.0f / Dn) - mu * mu;
  float inv = rsqrtf(var + 1e-5f);
#pragma unroll
  for (int t = 0; t < 4; ++t) {
    int c = tid + t * 256;
    o[c] = (v[t] - mu) * inv * sc[c] + bs[c];
  }
}

// ------------------------------ Gather / RoPE ------------------------------
__global__ void k_gather(const float* __restrict__ emb, const int* __restrict__ idx,
                         float* __restrict__ out, int nrows)
{
  size_t t = (size_t)blockIdx.x * 256 + threadIdx.x;
  if (t >= (size_t)nrows * Dn) return;
  int row = (int)(t >> 10);
  int col = (int)(t & 1023);
  out[t] = emb[(size_t)idx[row] * Dn + col];
}

__global__ void k_rope(float* __restrict__ x)   // (B*S, H*DK), pos = s, half = 32
{
  int t = blockIdx.x * 256 + threadIdx.x;
  if (t >= Bn * Sn * Hn * 32) return;
  int d   = t & 31;
  int hh  = (t >> 5) & 15;
  int row = t >> 9;                      // b*S + s
  int s   = row & (Sn - 1);
  float theta = __expf(-(float)d * (9.210340371976184f / 32.0f)); // 10000^(-d/32)
  float fr = (float)s * theta;
  float cs = cosf(fr), sn = sinf(fr);
  float* p = x + (size_t)row * Dn + hh * DKn;
  float x1 = p[d], x2 = p[d + 32];
  p[d]      = x1 * cs - x2 * sn;
  p[d + 32] = x1 * sn + x2 * cs;
}

// ------------------------- Self-attention (causal) -------------------------
// One block per (b, head, query i). Scores in LDS, residual fused:
// h[b,i,head*64..] += softmax(qK^T/8) V.
__global__ __launch_bounds__(256)
void k_self_attn(const float* __restrict__ qb, const float* __restrict__ kb,
                 const float* __restrict__ vb, float* __restrict__ h)
{
  const int idx = blockIdx.x;
  const int i  = idx & (Sn - 1);
  const int hh = (idx >> 10) & (Hn - 1);
  const int b  = idx >> 14;
  const int tid = threadIdx.x;

  __shared__ float qrow[DKn];
  __shared__ float sc[Sn];
  __shared__ float red[256];

  const float* qp = qb + ((size_t)(b * Sn + i)) * Dn + hh * DKn;
  if (tid < DKn) qrow[tid] = qp[tid];
  __syncthreads();

  // scores (causal), float4 dots
  const float4* q4 = (const float4*)qrow;
  for (int j = tid; j < Sn; j += 256) {
    float s = -3.0e38f;
    if (j <= i) {
      const float4* kp4 = (const float4*)(kb + ((size_t)(b * Sn + j)) * Dn + hh * DKn);
      float acc = 0.f;
#pragma unroll
      for (int d4 = 0; d4 < 16; ++d4) {
        float4 kv = kp4[d4]; float4 qv = q4[d4];
        acc += qv.x * kv.x + qv.y * kv.y + qv.z * kv.z + qv.w * kv.w;
      }
      s = acc * 0.125f;
    }
    sc[j] = s;
  }
  __syncthreads();

  // row max
  float m = -3.0e38f;
  for (int j = tid; j < Sn; j += 256) m = fmaxf(m, sc[j]);
  red[tid] = m; __syncthreads();
  for (int off = 128; off > 0; off >>= 1) {
    if (tid < off) red[tid] = fmaxf(red[tid], red[tid + off]);
    __syncthreads();
  }
  float mx = red[0];
  __syncthreads();

  // exp + sum
  float sum = 0.f;
  for (int j = tid; j < Sn; j += 256) {
    float e = __expf(sc[j] - mx);
    sc[j] = e;
    sum += e;
  }
  red[tid] = sum; __syncthreads();
  for (int off = 128; off > 0; off >>= 1) {
    if (tid < off) red[tid] += red[tid + off];
    __syncthreads();
  }
  float inv = 1.0f / red[0];
  __syncthreads();

  // out = P V  (4 groups x 64 dims)
  const int d = tid & 63, g = tid >> 6;
  float acc = 0.f;
  for (int j = g; j <= i; j += 4)
    acc += sc[j] * vb[((size_t)(b * Sn + j)) * Dn + hh * DKn + d];
  red[tid] = acc; __syncthreads();
  if (g == 0) {
    float o = (red[d] + red[64 + d] + red[128 + d] + red[192 + d]) * inv;
    float* hp = h + ((size_t)(b * Sn + i)) * Dn + hh * DKn;
    hp[d] = hp[d] + o;                // fused residual
  }
}

// --------------------- Chunked cross-attention (CCA) -----------------------
// One block per (b, chunk, l, head). 256 keys (NEIGH*NLEN), no mask.
__global__ __launch_bounds__(256)
void k_cca_attn(const float* __restrict__ qb, const float* __restrict__ kb,
                const float* __restrict__ vb, float* __restrict__ out)
{
  const int idx = blockIdx.x;
  const int hh = idx & 15;
  const int l  = (idx >> 4) & 63;
  const int c  = (idx >> 10) & 15;
  const int b  = idx >> 14;
  const int tid = threadIdx.x;

  __shared__ float qrow[DKn];
  __shared__ float prob[256];
  __shared__ float red[256];

  const float* qp = qb + ((size_t)(b * 1024 + c * CLn + l)) * Dn + hh * DKn;
  if (tid < DKn) qrow[tid] = qp[tid];
  __syncthreads();

  const size_t kvbase = ((size_t)((b * CHUNKSn + c) * 256)) * Dn + hh * DKn;
  const float4* kp4 = (const float4*)(kb + kvbase + (size_t)tid * Dn);
  const float4* q4  = (const float4*)qrow;
  float s = 0.f;
#pragma unroll
  for (int d4 = 0; d4 < 16; ++d4) {
    float4 kv = kp4[d4]; float4 qv = q4[d4];
    s += qv.x * kv.x + qv.y * kv.y + qv.z * kv.z + qv.w * kv.w;
  }
  s *= 0.125f;

  red[tid] = s; __syncthreads();
  for (int off = 128; off > 0; off >>= 1) {
    if (tid < off) red[tid] = fmaxf(red[tid], red[tid + off]);
    __syncthreads();
  }
  float mx = red[0];
  __syncthreads();
  float e = __expf(s - mx);
  red[tid] = e; __syncthreads();
  for (int off = 128; off > 0; off >>= 1) {
    if (tid < off) red[tid] += red[tid + off];
    __syncthreads();
  }
  float inv = 1.0f / red[0];
  prob[tid] = e * inv;
  __syncthreads();

  const int d = tid & 63, g = tid >> 6;
  float acc = 0.f;
  for (int j = g; j < 256; j += 4)
    acc += prob[j] * vb[kvbase + (size_t)j * Dn + d];
  red[tid] = acc; __syncthreads();
  if (g == 0)
    out[((size_t)(b * 1024 + c * CLn + l)) * Dn + hh * DKn + d] =
        red[d] + red[64 + d] + red[128 + d] + red[192 + d];
}

// h[b, 63+r, :] += proj[b, r, :]   for r in [0, 961)
__global__ void k_shift_add(float* __restrict__ h, const float* __restrict__ proj)
{
  const int RROWS = Sn - (CLn - 1);    // 961
  size_t t = (size_t)blockIdx.x * 256 + threadIdx.x;
  if (t >= (size_t)Bn * RROWS * Dn) return;
  int col = (int)(t & 1023);
  int r   = (int)((t >> 10) % RROWS);
  int b   = (int)((t >> 10) / RROWS);
  h[((size_t)(b * Sn + CLn - 1 + r)) * Dn + col] +=
      proj[((size_t)(b * 1024 + r)) * Dn + col];
}

// ------------------------------- Launcher ----------------------------------
static inline void gemm(const float* A, const float* Bm, const float* bias,
                        const float* Res, float* C, int M, int N, int K,
                        int relu, hipStream_t stream)
{
  dim3 grid(N / BN, M / BM);
  k_gemm_bf16<<<grid, 256, 0, stream>>>(A, Bm, bias, Res, C, M, N, K, relu);
}

extern "C" void kernel_launch(void* const* d_in, const int* in_sizes, int n_in,
                              void* d_out, int out_size, void* d_ws, size_t ws_size,
                              hipStream_t stream)
{
  (void)in_sizes; (void)n_in; (void)out_size; (void)ws_size;

  const float* emb     = (const float*)d_in[0];
  const float* sa_ln_s = (const float*)d_in[1];
  const float* sa_ln_b = (const float*)d_in[2];
  const float* Wq      = (const float*)d_in[3];
  const float* Wk      = (const float*)d_in[4];
  const float* Wv      = (const float*)d_in[5];
  const float* bq      = (const float*)d_in[6];
  const float* bk      = (const float*)d_in[7];
  const float* bv      = (const float*)d_in[8];
  const float* ff_ln_s = (const float*)d_in[9];
  const float* ff_ln_b = (const float*)d_in[10];
  const float* W1      = (const float*)d_in[11];
  const float* b1      = (const float*)d_in[12];
  const float* W2      = (const float*)d_in[13];
  const float* b2      = (const float*)d_in[14];
  const float* cln_s   = (const float*)d_in[15];
  const float* cln_b   = (const float*)d_in[16];
  const float* cWq     = (const float*)d_in[17];
  const float* cWk     = (const float*)d_in[18];
  const float* cWv     = (const float*)d_in[19];
  const float* cbq     = (const float*)d_in[20];
  const float* cbk     = (const float*)d_in[21];
  const float* cbv     = (const float*)d_in[22];
  const float* cWo     = (const float*)d_in[23];
  const float* cbo     = (const float*)d_in[24];
  const float* ne_s    = (const float*)d_in[25];
  const float* ne_b    = (const float*)d_in[26];
  const float* Wr      = (const float*)d_in[27];
  const float* br      = (const float*)d_in[28];
  const int*   x       = (const int*)d_in[29];
  const int*   ret     = (const int*)d_in[30];
  float* out = (float*)d_out;

  // Workspace layout (floats). Total = 5*2M + 3*8M = 35,651,584 f32 (~143MB).
  const size_t R  = (size_t)Bn * Sn;          // 2048 token rows
  const size_t RE = (size_t)Bn * CHUNKSn * NEIGHn * NLENn; // 8192 retrieval rows
  float* ws  = (float*)d_ws;
  float* h    = ws;                 // R*Dn
  float* hn   = h   + R * Dn;       // R*Dn   (also CCA hs_pad)
  float* q    = hn  + R * Dn;       // R*Dn   (also CCA q)
  float* kk   = q   + R * Dn;       // R*Dn   (also CCA proj)
  float* vv   = kk  + R * Dn;       // R*Dn   (also CCA attn out)
  float* big0 = vv  + R * Dn;       // RE*Dn  (ffn mid  / CCA K)
  float* big1 = big0 + RE * Dn;     // RE*Dn  (retrieval embeddings e)
  float* big2 = big1 + RE * Dn;     // RE*Dn  (CCA V)

  const int HD = Hn * DKn;          // 1024 == Dn

  // h = emb[x]
  k_gather<<<(unsigned)((R * Dn + 255) / 256), 256, 0, stream>>>(emb, x, h, (int)R);

  for (int p = 0; p < Ln; ++p) {
    // ---- self attention ----
    k_ln<<<(unsigned)R, 256, 0, stream>>>(h, hn, sa_ln_s + (size_t)p * Dn, sa_ln_b + (size_t)p * Dn);
    gemm(hn, Wq + (size_t)p * Dn * HD, bq + (size_t)p * HD, nullptr, q,  (int)R, HD, Dn, 0, stream);
    gemm(hn, Wk + (size_t)p * Dn * HD, bk + (size_t)p * HD, nullptr, kk, (int)R, HD, Dn, 0, stream);
    gemm(hn, Wv + (size_t)p * Dn * HD, bv + (size_t)p * HD, nullptr, vv, (int)R, HD, Dn, 0, stream);
    k_rope<<<(unsigned)((R * Hn * 32) / 256), 256, 0, stream>>>(q);
    k_rope<<<(unsigned)((R * Hn * 32) / 256), 256, 0, stream>>>(kk);
    k_self_attn<<<Bn * Hn * Sn, 256, 0, stream>>>(q, kk, vv, h);   // h += attn

    // ---- chunked cross attention at layer 3 ----
    if (p == CA_LAYERn) {
      k_gather<<<(unsigned)((RE * Dn + 255) / 256), 256, 0, stream>>>(emb, ret, big1, (int)RE);
      k_ln<<<(unsigned)RE, 256, 0, stream>>>(big1, big1, ne_s, ne_b);     // e = LN(ret_emb)
      k_cca_ln_pad<<<Bn * 1024, 256, 0, stream>>>(h, hn, cln_s, cln_b);   // hs_pad
      gemm(hn,   cWq, cbq, nullptr, q,    (int)R,  HD, Dn, 0, stream);    // cq
      gemm(big1, cWk, cbk, nullptr, big0, (int)RE, HD, Dn, 0, stream);    // cK
      gemm(big1, cWv, cbv, nullptr, big2, (int)RE, HD, Dn, 0, stream);    // cV
      k_cca_attn<<<Bn * CHUNKSn * CLn * Hn, 256, 0, stream>>>(q, big0, big2, vv);
      gemm(vv, cWo, cbo, nullptr, kk, (int)R, Dn, HD, 0, stream);         // proj
      k_shift_add<<<(unsigned)(((size_t)Bn * (Sn - CLn + 1) * Dn + 255) / 256), 256, 0, stream>>>(h, kk);
    }

    // ---- feed forward ----
    k_ln<<<(unsigned)R, 256, 0, stream>>>(h, hn, ff_ln_s + (size_t)p * Dn, ff_ln_b + (size_t)p * Dn);
    gemm(hn,   W1 + (size_t)p * Dn * DFFn, b1 + (size_t)p * DFFn, nullptr, big0, (int)R, DFFn, Dn, 1, stream);
    gemm(big0, W2 + (size_t)p * DFFn * Dn, b2 + (size_t)p * Dn,   h,       h,    (int)R, Dn, DFFn, 0, stream);
  }

  // logits = h @ Wr + br
  gemm(h, Wr, br, nullptr, out, (int)R, Vn, Dn, 0, stream);
}
